// MeanImputation_7842610282638
// MI455X (gfx1250) — matrix operand order repaired and checked by reading, not compiled, not run on hardware
//
#include <hip/hip_runtime.h>
#include <stdint.h>

#define N_NODES  50000
#define N_GENES  512
#define N_NEIGHS 16
#define HIDDEN   1024
#define NUM_DROP 51
#define KDIM     1024   // 2*N_GENES == HIDDEN, K for both GEMMs

typedef __attribute__((ext_vector_type(16))) __bf16 v16bf;
typedef __attribute__((ext_vector_type(8)))  float  v8f;

__device__ __forceinline__ unsigned short f2bf(float f) {
  unsigned u = __builtin_bit_cast(unsigned, f);
  u += 0x7FFFu + ((u >> 16) & 1u);          // round-to-nearest-even
  return (unsigned short)(u >> 16);
}
__device__ __forceinline__ float bf2f(unsigned short b) {
  return __builtin_bit_cast(float, ((unsigned)b) << 16);
}

// gfx1250 async global->LDS copy (16B per lane), tracked by ASYNCcnt.
__device__ __forceinline__ void asyncLoad16B(unsigned ldsAddr, unsigned long long gaddr) {
  asm volatile("global_load_async_to_lds_b128 %0, %1, off"
               :: "v"(ldsAddr), "v"(gaddr) : "memory");
}
__device__ __forceinline__ void waitAsync0() {
  asm volatile("s_wait_asynccnt 0x0" ::: "memory");
}

// ---------------------------------------------------------------------------
// 0) Convert W1 [1024x1024] and W2 [512x1024] fp32 -> bf16 in workspace
// ---------------------------------------------------------------------------
__global__ __launch_bounds__(256)
void convert_w_kernel(const float* __restrict__ W1, const float* __restrict__ W2,
                      unsigned short* __restrict__ W1b, unsigned short* __restrict__ W2b) {
  int id = blockIdx.x * 256 + threadIdx.x;
  if (id < HIDDEN * KDIM)  W1b[id] = f2bf(W1[id]);
  if (id < N_GENES * KDIM) W2b[id] = f2bf(W2[id]);
}

// ---------------------------------------------------------------------------
// 1) Per-node drop mask + bf16 pack + x_init gather.  1 block / node.
// ---------------------------------------------------------------------------
__global__ __launch_bounds__(256)
void prep_kernel(const float* __restrict__ x, const int* __restrict__ dropIdx,
                 unsigned short* __restrict__ hidden, float* __restrict__ outInit) {
  __shared__ int sdrop[NUM_DROP];
  const int n = blockIdx.x;
  const int t = threadIdx.x;
  if (t < NUM_DROP) sdrop[t] = dropIdx[n * NUM_DROP + t];
  __syncthreads();
  const float* xr = x + (size_t)n * N_GENES;
  unsigned short* hr = hidden + (size_t)n * KDIM;
  #pragma unroll
  for (int gg = 0; gg < 2; ++gg) {
    int g = t + gg * 256;
    float v = xr[g];
    int lo = 0, hi = NUM_DROP - 1;        // binary search in sorted drop list
    bool dropped = false;
    while (lo <= hi) {
      int mid = (lo + hi) >> 1;
      int dv = sdrop[mid];
      if (dv == g) { dropped = true; break; }
      if (dv < g) lo = mid + 1; else hi = mid - 1;
    }
    hr[g] = f2bf(dropped ? 0.0f : v);
  }
  if (t < NUM_DROP) outInit[(size_t)n * NUM_DROP + t] = xr[sdrop[t]];
}

// ---------------------------------------------------------------------------
// 2) 16-neighbor mean pool (bf16 in/out), writes hidden[:,512:1024].
// ---------------------------------------------------------------------------
__global__ __launch_bounds__(256)
void niche_kernel(const int* __restrict__ nodeIdx, unsigned short* __restrict__ hidden) {
  __shared__ int snb[N_NEIGHS];
  const int n = blockIdx.x;
  const int t = threadIdx.x;
  if (t < N_NEIGHS) snb[t] = nodeIdx[n * N_NEIGHS + t];
  __syncthreads();
  const int g2 = t * 2;                       // 512 genes / 256 threads
  float acc0 = 0.f, acc1 = 0.f;
  #pragma unroll
  for (int j = 0; j < N_NEIGHS; ++j) {
    const unsigned pk = *(const unsigned*)&hidden[(size_t)snb[j] * KDIM + g2];
    acc0 += bf2f((unsigned short)(pk & 0xFFFFu));
    acc1 += bf2f((unsigned short)(pk >> 16));
  }
  unsigned out = ((unsigned)f2bf(acc1 * (1.0f / N_NEIGHS)) << 16) |
                  (unsigned)f2bf(acc0 * (1.0f / N_NEIGHS));
  *(unsigned*)&hidden[(size_t)n * KDIM + N_GENES + g2] = out;
}

// ---------------------------------------------------------------------------
// 3) Tiled bf16 WMMA GEMM with double-buffered async global->LDS staging.
//    Block tile 128x128, 8 waves x (2x4) 16x16x32 WMMA tiles, K-step 32.
// ---------------------------------------------------------------------------
#define TILE_M 128
#define TILE_N 128
#define TILE_K 32
#define LDSS   40      // padded LDS row stride (elements)

template<int NOUT, bool RELU, bool OUTBF16>
__global__ __launch_bounds__(256)
void gemm_bf16_wmma(const unsigned short* __restrict__ A,
                    const unsigned short* __restrict__ W,
                    const float* __restrict__ bias,
                    unsigned short* __restrict__ outB,
                    float* __restrict__ outF,
                    int nRows, int mBlocks) {
  __shared__ unsigned short As[2][TILE_M * LDSS];
  __shared__ unsigned short Ws[2][TILE_N * LDSS];

  const int t        = threadIdx.x;
  const int lane     = t & 31;
  const int wave     = t >> 5;
  const int lane16   = lane & 15;
  const int laneHalf = lane >> 4;          // which half-wave
  const int bm = blockIdx.x % mBlocks;
  const int bn = blockIdx.x / mBlocks;
  const int mBase = bm * TILE_M;
  const int nBase = bn * TILE_N;
  const int wm = wave & 3;                 // M-group: rows 32*wm .. +31
  const int wn = wave >> 2;                // N-group: cols 64*wn .. +63

  v8f acc[2][4] = {};

  // cooperative staging: 2 threads per row, 32B (= two b128 async ops) each
  const int ldRow = t >> 1;
  const int ldCol = (t & 1) * 16;          // element offset (0 or 16)
  int aRow = mBase + ldRow; if (aRow >= nRows) aRow = nRows - 1;  // clamp; stores guarded
  const unsigned long long aSrc =
      (unsigned long long)(uintptr_t)(A + (size_t)aRow * KDIM + ldCol);
  const unsigned long long wSrc =
      (unsigned long long)(uintptr_t)(W + (size_t)(nBase + ldRow) * KDIM + ldCol); // NOUT%128==0
  const unsigned ldsA0 = (unsigned)(uintptr_t)&As[0][ldRow * LDSS + ldCol];
  const unsigned ldsW0 = (unsigned)(uintptr_t)&Ws[0][ldRow * LDSS + ldCol];
  const unsigned bufA  = (unsigned)(sizeof(unsigned short) * TILE_M * LDSS);

  // preload K-tile 0 into buffer 0
  asyncLoad16B(ldsA0,      aSrc);
  asyncLoad16B(ldsA0 + 16, aSrc + 16);
  asyncLoad16B(ldsW0,      wSrc);
  asyncLoad16B(ldsW0 + 16, wSrc + 16);

  int buf = 0;
  for (int k = 0; k < KDIM; k += TILE_K) {
    waitAsync0();        // my async writes for buf are in LDS
    __syncthreads();     // everyone's buf writes done; everyone done with buf^1
    if (k + TILE_K < KDIM) {
      const unsigned nb = (buf ^ 1) ? bufA : 0u;
      const unsigned long long gk = (unsigned long long)(k + TILE_K) * 2ull;
      asyncLoad16B(ldsA0 + nb,      aSrc + gk);
      asyncLoad16B(ldsA0 + nb + 16, aSrc + gk + 16);
      asyncLoad16B(ldsW0 + nb,      wSrc + gk);
      asyncLoad16B(ldsW0 + nb + 16, wSrc + gk + 16);
    }

    const unsigned short* Ab = &As[buf][0];
    const unsigned short* Wb = &Ws[buf][0];

    // A fragments (16x32): lanes 0-15 hold K 0-15, lanes 16-31 hold K 16-31
    v16bf afr[2];
    #pragma unroll
    for (int i = 0; i < 2; ++i) {
      const int row = wm * 32 + i * 16 + lane16;
      const int klo = laneHalf * 16;
      uint4* ap = (uint4*)&afr[i];
      ap[0] = *(const uint4*)&Ab[row * LDSS + klo];
      ap[1] = *(const uint4*)&Ab[row * LDSS + klo + 8];
    }
    // B fragments (32x16): lanes 0-15 hold K 0-15, lanes 16-31 hold K 16-31 (col = lane%16)
    v16bf bfr[4];
    #pragma unroll
    for (int j = 0; j < 4; ++j) {
      const int col = wn * 64 + j * 16 + lane16;
      const int kb  = laneHalf * 16;
      uint4* bp = (uint4*)&bfr[j];
      bp[0] = *(const uint4*)&Wb[col * LDSS + kb];
      bp[1] = *(const uint4*)&Wb[col * LDSS + kb + 8];
    }
    #pragma unroll
    for (int i = 0; i < 2; ++i)
      #pragma unroll
      for (int j = 0; j < 4; ++j)
        acc[i][j] = __builtin_amdgcn_wmma_f32_16x16x32_bf16(
            false, afr[i], false, bfr[j], (short)0, acc[i][j], false, false);

    buf ^= 1;
  }

  // Epilogue: lane holds col = 16*tile + lane%16, rows r + 8*(lane/16)
  const bool full = (mBase + TILE_M) <= nRows;
  #pragma unroll
  for (int j = 0; j < 4; ++j) {
    const int col = nBase + wn * 64 + j * 16 + lane16;
    const float bv = bias[col];
    #pragma unroll
    for (int i = 0; i < 2; ++i) {
      const int rowT = mBase + wm * 32 + i * 16 + laneHalf * 8;
      if (full) {
        #pragma unroll
        for (int r = 0; r < 8; ++r) {
          float v = acc[i][j][r] + bv;
          if (RELU) v = v > 0.f ? v : 0.f;
          if (OUTBF16) outB[(size_t)(rowT + r) * NOUT + col] = f2bf(v);
          else         outF[(size_t)(rowT + r) * NOUT + col] = v;
        }
      } else {
        #pragma unroll
        for (int r = 0; r < 8; ++r) {
          const int row = rowT + r;
          if (row < nRows) {
            float v = acc[i][j][r] + bv;
            if (RELU) v = v > 0.f ? v : 0.f;
            if (OUTBF16) outB[(size_t)row * NOUT + col] = f2bf(v);
            else         outF[(size_t)row * NOUT + col] = v;
          }
        }
      }
    }
  }
}

// ---------------------------------------------------------------------------
// 4) Gather recon at dropped columns -> x_recon
// ---------------------------------------------------------------------------
__global__ __launch_bounds__(256)
void gather_kernel(const float* __restrict__ recon, const int* __restrict__ dropIdx,
                   float* __restrict__ outRecon) {
  int id = blockIdx.x * 256 + threadIdx.x;
  if (id >= N_NODES * NUM_DROP) return;
  int n = id / NUM_DROP;
  int g = dropIdx[id];
  outRecon[id] = recon[(size_t)n * N_GENES + g];
}

// ---------------------------------------------------------------------------
extern "C" void kernel_launch(void* const* d_in, const int* in_sizes, int n_in,
                              void* d_out, int out_size, void* d_ws, size_t ws_size,
                              hipStream_t stream) {
  const float* x   = (const float*)d_in[0];
  const float* W1  = (const float*)d_in[1];
  const float* b1  = (const float*)d_in[2];
  const float* W2  = (const float*)d_in[3];
  const float* b2  = (const float*)d_in[4];
  const int* dropIdx = (const int*)d_in[5];
  const int* nodeIdx = (const int*)d_in[6];

  char* ws = (char*)d_ws;
  size_t off = 0;
  unsigned short* hidden = (unsigned short*)(ws + off); off += (size_t)N_NODES * KDIM * 2;   // [N,1024] bf16
  unsigned short* W1b    = (unsigned short*)(ws + off); off += (size_t)HIDDEN * KDIM * 2;    // bf16
  unsigned short* W2b    = (unsigned short*)(ws + off); off += (size_t)N_GENES * KDIM * 2;   // bf16
  unsigned short* hbuf   = (unsigned short*)(ws + off); off += (size_t)N_NODES * HIDDEN * 2; // [N,1024] bf16
  float*          recon  = (float*)(ws + off);          off += (size_t)N_NODES * N_GENES * 4;// [N,512] f32

  float* outInit  = (float*)d_out;
  float* outRecon = outInit + (size_t)N_NODES * NUM_DROP;

  convert_w_kernel<<<dim3((HIDDEN * KDIM + 255) / 256), dim3(256), 0, stream>>>(W1, W2, W1b, W2b);
  prep_kernel<<<dim3(N_NODES), dim3(256), 0, stream>>>(x, dropIdx, hidden, outInit);
  niche_kernel<<<dim3(N_NODES), dim3(256), 0, stream>>>(nodeIdx, hidden);

  const int mBlocks = (N_NODES + TILE_M - 1) / TILE_M;  // 391
  gemm_bf16_wmma<HIDDEN, true, true><<<dim3(mBlocks * (HIDDEN / TILE_N)), dim3(256), 0, stream>>>(
      hidden, W1b, b1, hbuf, (float*)nullptr, N_NODES, mBlocks);
  gemm_bf16_wmma<N_GENES, false, false><<<dim3(mBlocks * (N_GENES / TILE_N)), dim3(256), 0, stream>>>(
      hbuf, W2b, b2, (unsigned short*)nullptr, recon, N_NODES, mBlocks);

  gather_kernel<<<dim3((N_NODES * NUM_DROP + 255) / 256), dim3(256), 0, stream>>>(
      recon, dropIdx, outRecon);
}
